// SimpleLlamaAttention_68788196213261
// MI455X (gfx1250) — matrix operand order repaired and checked by reading, not compiled, not run on hardware
//
#include <hip/hip_runtime.h>
#include <hip/hip_bf16.h>

// ---- problem constants (match reference) ----
constexpr int BB  = 2;
constexpr int SS  = 2048;
constexpr int HID = 4096;
constexpr int HC  = 32;
constexpr int KVC = 8;
constexpr int DC  = 128;
constexpr int NQKV = HC * DC + 2 * KVC * DC;   // 6144
constexpr float SCALE = 0.08838834764831845f;  // 1/sqrt(128)
constexpr int KD  = 64;                        // staged K depth per GEMM buffer

typedef __attribute__((ext_vector_type(16))) _Float16 v16h;
typedef __attribute__((ext_vector_type(8)))  _Float16 v8h;
typedef __attribute__((ext_vector_type(8)))  float    v8f;
typedef unsigned int u32x4 __attribute__((ext_vector_type(4)));
typedef int          i32x8 __attribute__((ext_vector_type(8)));
typedef int          i32x4 __attribute__((ext_vector_type(4)));

static __device__ inline v8f wmma_f16(v16h a, v16h b, v8f c) {
  return __builtin_amdgcn_wmma_f32_16x16x32_f16(
      false, a, false, b, (short)0, c, false, false);
}

static __device__ inline v16h pack16(v8h lo, v8h hi) {
  return __builtin_shufflevector(lo, hi, 0,1,2,3,4,5,6,7,8,9,10,11,12,13,14,15);
}

// A-fragment (16x32, M x K): lane m(+16*khalf) holds
// [k0 + 16*khalf .. +15] of row m (contiguous).
static __device__ inline v16h load_a(const _Float16* rowp, int koffA) {
  v8h lo = *(const v8h*)(rowp + 2 * koffA);
  v8h hi = *(const v8h*)(rowp + 2 * koffA + 8);
  return pack16(lo, hi);
}

// ---- Tensor Data Mover: 2-D tile (tile_d1 rows x tile_d0 contiguous) -> LDS ----
// data_size = 2 bytes. Dims/strides in elements. See CDNA5 ISA.
static __device__ inline void tdm_load_2d(unsigned lds_addr, const void* gptr,
                                          unsigned tensor_d0, unsigned tensor_d1,
                                          unsigned tile_d0, unsigned tile_d1,
                                          unsigned long long stride0) {
  unsigned long long ga = (unsigned long long)gptr;
  u32x4 g0;
  g0[0] = 1u;                                   // count=1, user descriptor
  g0[1] = lds_addr;                             // LDS byte address
  g0[2] = (unsigned)ga;                         // global_addr[31:0]
  g0[3] = (unsigned)((ga >> 32) & 0x1FFFFFFu) | (2u << 30);  // addr[56:32], type=2
  i32x8 g1;
  g1[0] = (int)(1u << 16);                      // workgroup_mask=0, data_size=1 (2B)
  g1[1] = (int)((tensor_d0 & 0xFFFFu) << 16);   // tensor_dim0[15:0] @ bits 63:48
  g1[2] = (int)(((tensor_d0 >> 16) & 0xFFFFu) | ((tensor_d1 & 0xFFFFu) << 16));
  g1[3] = (int)(((tensor_d1 >> 16) & 0xFFFFu) | ((tile_d0 & 0xFFFFu) << 16));
  g1[4] = (int)(tile_d1 & 0xFFFFu);             // tile_dim1; tile_dim2=0
  g1[5] = (int)(unsigned)(stride0 & 0xFFFFFFFFu);
  g1[6] = (int)(unsigned)((stride0 >> 32) & 0xFFFFu);  // stride[47:32]; dim1_stride=0
  g1[7] = 0;
  i32x4 z4 = {0, 0, 0, 0};
#if __clang_major__ >= 23
  i32x8 z8 = {};
  __builtin_amdgcn_tensor_load_to_lds(g0, g1, z4, z4, z8, 0);
#else
  __builtin_amdgcn_tensor_load_to_lds(g0, g1, z4, z4, 0);
#endif
}

static __device__ inline unsigned lds_byte_addr(const void* p) {
  return (unsigned)(unsigned long long)p;   // LDS aperture: low 32 bits = LDS offset
}

// ---------------- fp32 -> f16 convert ----------------
__global__ void cvt_f32_to_f16(const float* __restrict__ src,
                               _Float16* __restrict__ dst, size_t n) {
  size_t i = (size_t)blockIdx.x * blockDim.x + threadIdx.x;
  size_t stride = (size_t)gridDim.x * blockDim.x;
  for (; i < n; i += stride) dst[i] = (_Float16)src[i];
}

// ---------------- fused QKV projection + RoPE ----------------
// block = 128 (4 waves), tile 64(M) x 128(N). Weight tile (128N x 64K, 16KB)
// TDM-staged in LDS, double-buffered, shared by all 4 waves. 16 WMMAs per
// wave between barriers; A fragments stream from global (L2-resident X).
__global__ void __launch_bounds__(128)
qkv_rope_kernel(const _Float16* __restrict__ X,
                const _Float16* __restrict__ wq,
                const _Float16* __restrict__ wk,
                const _Float16* __restrict__ wv,
                const float* __restrict__ cosp,
                const float* __restrict__ sinp,
                _Float16* __restrict__ qo,   // [B,H,S,D]
                _Float16* __restrict__ ko,   // [B,KV,S,D]
                _Float16* __restrict__ vto)  // [B,KV,D,S]
{
  __shared__ __align__(64) _Float16 ldsW[2][128 * KD];   // 2 x 16KB weight tiles

  const int lane  = threadIdx.x & 31;
  const int wid   = threadIdx.x >> 5;
  const int m0    = blockIdx.x * 64 + wid * 16;
  const int n0    = blockIdx.y * 128;
  const int nB    = lane & 15;
  const int half  = lane >> 4;
  const int koffA = half * 8;

  const _Float16* W;
  int nloc, region;
  if (n0 < HC * DC)                 { W = wq; nloc = n0;            region = 0; }
  else if (n0 < HC * DC + KVC * DC) { W = wk; nloc = n0 - HC * DC;  region = 1; }
  else                              { W = wv; nloc = n0 - HC*DC - KVC*DC; region = 2; }

  v8f acc[8] = {};
  const _Float16* xrow = X + (size_t)(m0 + (lane & 15)) * HID;
  const _Float16* wbase = W + (size_t)nloc * HID;

  if (wid == 0) {
    tdm_load_2d(lds_byte_addr(&ldsW[0][0]), wbase, KD, 128, KD, 128, HID);
    __builtin_amdgcn_s_wait_tensorcnt(0);
  }
  __syncthreads();

  int cur = 0;
  for (int k0 = 0; k0 < HID; k0 += KD) {
    if (wid == 0 && k0 + KD < HID)
      tdm_load_2d(lds_byte_addr(&ldsW[cur ^ 1][0]), wbase + k0 + KD,
                  KD, 128, KD, 128, HID);

#pragma unroll
    for (int ks = 0; ks < KD / 32; ++ks) {
      v16h afrag = load_a(xrow + k0 + ks * 32, koffA);
#pragma unroll
      for (int nt = 0; nt < 8; ++nt) {
        v16h bfrag = *(const v16h*)(
            &ldsW[cur][(nt * 16 + nB) * KD + ks * 32 + half * 16]);
        acc[nt] = wmma_f16(afrag, bfrag, acc[nt]);
      }
    }

    __syncthreads();                              // all waves done with ldsW[cur]
    if (wid == 0) __builtin_amdgcn_s_wait_tensorcnt(0);
    __syncthreads();                              // next buffer published
    cur ^= 1;
  }

  // epilogue: C layout -> lanes 0-15 rows r, lanes 16-31 rows r+8
#pragma unroll
  for (int nt = 0; nt < 8; ++nt) {
#pragma unroll
    for (int r = 0; r < 8; ++r) {
      float x   = acc[nt][r];
      int row   = m0 + r + half * 8;
      int b     = row >> 11;
      int s     = row & (SS - 1);
      int col   = nloc + nt * 16 + nB;
      if (region < 2) {                       // RoPE for q and k
        float partner = __shfl_xor(x, 1, 32);
        int   d  = col & (DC - 1);
        float c  = cosp[s * DC + d];
        float sn = sinp[s * DC + d];
        float rot = (d & 1) ? partner : -partner;
        x = x * c + rot * sn;
      }
      int hd = col >> 7, d = col & (DC - 1);
      if (region == 0)
        qo[(((size_t)b * HC + hd) * SS + s) * DC + d] = (_Float16)x;
      else if (region == 1)
        ko[(((size_t)b * KVC + hd) * SS + s) * DC + d] = (_Float16)x;
      else
        vto[(((size_t)b * KVC + hd) * DC + d) * SS + s] = (_Float16)x;
    }
  }
}

// ---------------- flash attention (causal, GQA) ----------------
// block = 128 (4 waves) covering 64 query rows; K/V 32-key tiles TDM-staged
// in LDS, double-buffered, shared by all 4 waves.
__global__ void __launch_bounds__(128)
flash_attn_kernel(const _Float16* __restrict__ q,  // [B,H,S,D]
                  const _Float16* __restrict__ k,  // [B,KV,S,D]
                  const _Float16* __restrict__ vt, // [B,KV,D,S]
                  _Float16* __restrict__ attn)     // [B,S,H*D]
{
  __shared__ __align__(64) _Float16 kT[2][32 * 128];   // keys x d     (8KB each)
  __shared__ __align__(64) _Float16 vT[2][128 * 32];   // d x keys     (8KB each)
  __shared__ __align__(64) _Float16 lds_p[4][16 * 32]; // per-wave P   (1KB each)

  const int lane  = threadIdx.x & 31;
  const int wid   = threadIdx.x >> 5;
  const int n     = lane & 15;
  const int half  = lane >> 4;
  const int koffA = half * 8;

  const int s0blk = blockIdx.x * 64;
  const int s0    = s0blk + wid * 16;        // this wave's 16 query rows
  const int bh = blockIdx.y;
  const int b  = bh / HC;
  const int h  = bh % HC;
  const int kv = h / (HC / KVC);

  const _Float16* qh  = q  + ((size_t)(b * HC + h) * SS) * DC;
  const _Float16* kh  = k  + ((size_t)(b * KVC + kv) * SS) * DC;
  const _Float16* vth = vt + ((size_t)(b * KVC + kv) * DC) * SS;

  // preload the 16x128 Q block as 4 A-fragments
  v16h qfrag[4];
  const _Float16* qrow = qh + (size_t)(s0 + (lane & 15)) * DC;
#pragma unroll
  for (int kk = 0; kk < 4; ++kk) qfrag[kk] = load_a(qrow + kk * 32, koffA);

  float m_run[8], l_run[8];
  v8f acc_o[8];
#pragma unroll
  for (int r = 0; r < 8; ++r) { m_run[r] = -3.0e38f; l_run[r] = 0.0f; }
#pragma unroll
  for (int nt = 0; nt < 8; ++nt) acc_o[nt] = v8f{};

  const int tmax = s0blk + 64;               // causal bound for the block

  if (wid == 0) {
    tdm_load_2d(lds_byte_addr(&kT[0][0]), kh,  128, 32, 128, 32, DC);
    tdm_load_2d(lds_byte_addr(&vT[0][0]), vth, 32, 128, 32, 128, SS);
    __builtin_amdgcn_s_wait_tensorcnt(0);
  }
  __syncthreads();

  int cur = 0;
  for (int t0 = 0; t0 < tmax; t0 += 32) {
    if (wid == 0 && t0 + 32 < tmax) {
      tdm_load_2d(lds_byte_addr(&kT[cur ^ 1][0]), kh + (size_t)(t0 + 32) * DC,
                  128, 32, 128, 32, DC);
      tdm_load_2d(lds_byte_addr(&vT[cur ^ 1][0]), vth + t0 + 32,
                  32, 128, 32, 128, SS);
    }

    if (t0 < s0 + 16) {     // this wave has unmasked columns in this chunk
      // ---- scores for two 16-key subtiles ----
      float sc[2][8];
#pragma unroll
      for (int j = 0; j < 2; ++j) {
        int tk = t0 + 16 * j;
        v8f acc_s = {};
        if (tk < s0 + 16) {
#pragma unroll
          for (int kk = 0; kk < 4; ++kk) {
            v16h bfrag = *(const v16h*)(
                &kT[cur][(j * 16 + n) * DC + kk * 32 + half * 16]);
            acc_s = wmma_f16(qfrag[kk], bfrag, acc_s);
          }
        }
#pragma unroll
        for (int r = 0; r < 8; ++r) {
          int row = s0 + r + half * 8;
          int col = tk + n;
          float v = acc_s[r] * SCALE;
          sc[j][r] = (tk < s0 + 16 && col <= row) ? v : -3.0e38f;
        }
      }

      // ---- online softmax (row reductions inside each 16-lane half) ----
      float alpha[8];
#pragma unroll
      for (int r = 0; r < 8; ++r) {
        float mt = fmaxf(sc[0][r], sc[1][r]);
#pragma unroll
        for (int mk = 1; mk <= 8; mk <<= 1) mt = fmaxf(mt, __shfl_xor(mt, mk, 32));
        float mn = fmaxf(m_run[r], mt);
        alpha[r] = __expf(m_run[r] - mn);
        m_run[r] = mn;
        float p0 = __expf(sc[0][r] - mn);
        float p1 = __expf(sc[1][r] - mn);
        float rs = p0 + p1;
#pragma unroll
        for (int mk = 1; mk <= 8; mk <<= 1) rs += __shfl_xor(rs, mk, 32);
        l_run[r] = l_run[r] * alpha[r] + rs;
        lds_p[wid][(r + half * 8) * 32 + n]      = (_Float16)p0;
        lds_p[wid][(r + half * 8) * 32 + 16 + n] = (_Float16)p1;
      }

      asm volatile("s_wait_dscnt 0" ::: "memory");  // per-wave P staging

      const _Float16* prow = &lds_p[wid][(lane & 15) * 32];
      v16h pfrag = load_a(prow, koffA);

      // ---- O = O*alpha + P @ V ----
#pragma unroll
      for (int nt = 0; nt < 8; ++nt) {
#pragma unroll
        for (int r = 0; r < 8; ++r) acc_o[nt][r] *= alpha[r];
        v16h bfrag = *(const v16h*)(&vT[cur][(nt * 16 + n) * 32 + half * 16]);
        acc_o[nt] = wmma_f16(pfrag, bfrag, acc_o[nt]);
      }
    }

    __syncthreads();                               // buffers consumed
    if (wid == 0) __builtin_amdgcn_s_wait_tensorcnt(0);
    __syncthreads();                               // next buffers published
    cur ^= 1;
  }

  // ---- normalize and store to [B, S, H*D] ----
#pragma unroll
  for (int r = 0; r < 8; ++r) {
    float inv = 1.0f / l_run[r];
    int row = s0 + r + half * 8;
#pragma unroll
    for (int nt = 0; nt < 8; ++nt) {
      int col = h * DC + nt * 16 + n;
      attn[((size_t)b * SS + row) * (HC * DC) + col] =
          (_Float16)(acc_o[nt][r] * inv);
    }
  }
}

// ---------------- output projection: out = attn @ wo.T (fp32 out) ----------------
// block = 128 (4 waves), tile 64(M) x 128(N), TDM-staged 128N x 64K weight tiles.
__global__ void __launch_bounds__(128)
oproj_kernel(const _Float16* __restrict__ A,   // [B*S, 4096] f16
             const _Float16* __restrict__ wo,  // [4096, 4096] f16
             float* __restrict__ out)          // [B*S, 4096] f32
{
  __shared__ __align__(64) _Float16 ldsW[2][128 * KD];

  const int lane  = threadIdx.x & 31;
  const int wid   = threadIdx.x >> 5;
  const int m0    = blockIdx.x * 64 + wid * 16;
  const int n0    = blockIdx.y * 128;
  const int nB    = lane & 15;
  const int half  = lane >> 4;
  const int koffA = half * 8;

  v8f acc[8] = {};
  const _Float16* arow  = A + (size_t)(m0 + (lane & 15)) * HID;
  const _Float16* wbase = wo + (size_t)n0 * HID;

  if (wid == 0) {
    tdm_load_2d(lds_byte_addr(&ldsW[0][0]), wbase, KD, 128, KD, 128, HID);
    __builtin_amdgcn_s_wait_tensorcnt(0);
  }
  __syncthreads();

  int cur = 0;
  for (int k0 = 0; k0 < HID; k0 += KD) {
    if (wid == 0 && k0 + KD < HID)
      tdm_load_2d(lds_byte_addr(&ldsW[cur ^ 1][0]), wbase + k0 + KD,
                  KD, 128, KD, 128, HID);

#pragma unroll
    for (int ks = 0; ks < KD / 32; ++ks) {
      v16h afrag = load_a(arow + k0 + ks * 32, koffA);
#pragma unroll
      for (int nt = 0; nt < 8; ++nt) {
        v16h bfrag = *(const v16h*)(
            &ldsW[cur][(nt * 16 + nB) * KD + ks * 32 + half * 16]);
        acc[nt] = wmma_f16(afrag, bfrag, acc[nt]);
      }
    }

    __syncthreads();
    if (wid == 0) __builtin_amdgcn_s_wait_tensorcnt(0);
    __syncthreads();
    cur ^= 1;
  }

#pragma unroll
  for (int nt = 0; nt < 8; ++nt) {
#pragma unroll
    for (int r = 0; r < 8; ++r) {
      int row = m0 + r + half * 8;
      int col = n0 + nt * 16 + nB;
      out[(size_t)row * HID + col] = acc[nt][r];
    }
  }
}

extern "C" void kernel_launch(void* const* d_in, const int* in_sizes, int n_in,
                              void* d_out, int out_size, void* d_ws, size_t ws_size,
                              hipStream_t stream) {
  const float* hidden = (const float*)d_in[0];   // [B,S,HID]
  const float* cosp   = (const float*)d_in[1];   // [S,D]
  const float* sinp   = (const float*)d_in[2];   // [S,D]
  const float* wq     = (const float*)d_in[3];   // [H*D, HID]
  const float* wk     = (const float*)d_in[4];   // [KV*D, HID]
  const float* wv     = (const float*)d_in[5];   // [KV*D, HID]
  const float* wo     = (const float*)d_in[6];   // [HID, H*D]
  float* out = (float*)d_out;

  const size_t nHid = (size_t)BB * SS * HID;        // 16,777,216
  const size_t nWq  = (size_t)HC * DC * HID;        // 16,777,216
  const size_t nWk  = (size_t)KVC * DC * HID;       //  4,194,304
  const size_t nQ   = (size_t)BB * HC * SS * DC;    // 16,777,216
  const size_t nK   = (size_t)BB * KVC * SS * DC;   //  4,194,304

  _Float16* wsp      = (_Float16*)d_ws;
  _Float16* hid_h    = wsp;                size_t off = nHid;
  _Float16* wq_h     = wsp + off;          off += nWq;
  _Float16* wk_h     = wsp + off;          off += nWk;
  _Float16* wv_h     = wsp + off;          off += nWk;
  _Float16* wo_h     = wsp + off;          off += nWq;
  _Float16* q_h      = wsp + off;          off += nQ;
  _Float16* k_h      = wsp + off;          off += nK;
  _Float16* vt_h     = wsp + off;          off += nK;
  _Float16* attn_h   = wsp + off;          off += nQ;
  (void)ws_size; (void)in_sizes; (void)n_in; (void)out_size;

  const int CVB = 256, CVG = 4096;
  cvt_f32_to_f16<<<CVG, CVB, 0, stream>>>(hidden, hid_h, nHid);
  cvt_f32_to_f16<<<CVG, CVB, 0, stream>>>(wq,     wq_h,  nWq);
  cvt_f32_to_f16<<<CVG, CVB, 0, stream>>>(wk,     wk_h,  nWk);
  cvt_f32_to_f16<<<CVG, CVB, 0, stream>>>(wv,     wv_h,  nWk);
  cvt_f32_to_f16<<<CVG, CVB, 0, stream>>>(wo,     wo_h,  nWq);

  dim3 gQKV((BB * SS) / 64, NQKV / 128);
  qkv_rope_kernel<<<gQKV, 128, 0, stream>>>(hid_h, wq_h, wk_h, wv_h,
                                            cosp, sinp, q_h, k_h, vt_h);

  dim3 gAtt(SS / 64, BB * HC);
  flash_attn_kernel<<<gAtt, 128, 0, stream>>>(q_h, k_h, vt_h, attn_h);

  dim3 gO((BB * SS) / 64, HID / 128);
  oproj_kernel<<<gO, 128, 0, stream>>>(attn_h, wo_h, out);
}
